// StreamVAEModel_57621281243425
// MI455X (gfx1250) — matrix operand order repaired and checked by reading, not compiled, not run on hardware
//
#include <hip/hip_runtime.h>
#include <math.h>

// ---------------------------------------------------------------------------
// StreamVAE forward for MI455X (gfx1250, wave32, WMMA).
// Dense projections -> fully-unrolled v_wmma_f32_16x16x32_f16 GEMM (K
// templated, 2 output tiles per wave sharing one A fragment); LSTM/EMA scans
// -> LDS sequential kernels; attention core -> flash-style online softmax.
// ---------------------------------------------------------------------------

typedef __attribute__((ext_vector_type(16))) _Float16 v16h;
typedef __attribute__((ext_vector_type(8)))  float    v8f;

#define B_  8
#define T_  1024
#define F_  64
#define L_  64
#define M_  (B_ * T_)   // 8192 rows for all GEMMs

// ---------------- device helpers ----------------

__device__ __forceinline__ float sigm(float x) { return 1.f / (1.f + __expf(-x)); }
__device__ __forceinline__ float softplusf(float x) { return x > 20.f ? x : log1pf(__expf(x)); }

// block-of-64 sum reduction (2 waves)
__device__ __forceinline__ float rowsum64(float v, float* red) {
  int t = threadIdx.x;
  red[t] = v;
  __syncthreads();
  if (t < 32) red[t] += red[t + 32];
  __syncthreads();
  if (t < 16) red[t] += red[t + 16];
  __syncthreads();
  if (t < 8)  red[t] += red[t + 8];
  __syncthreads();
  if (t < 4)  red[t] += red[t + 4];
  __syncthreads();
  if (t < 2)  red[t] += red[t + 2];
  __syncthreads();
  if (t < 1)  red[t] += red[t + 1];
  __syncthreads();
  float r = red[0];
  __syncthreads();
  return r;
}

// ---------------- WMMA GEMM: Y[M,N] = X[M,K] @ W[N,K]^T (+ b0 + b1) --------
// grid = (N/32, M/128), block = 256 (8 waves).
// Each wave computes a 16x32 slab: one A fragment, two B fragments, two WMMAs
// per unrolled K-step (K is a template constant -> full unroll).
//
// A fragment (16x32 f16): lanes 0-15 row M=l: K k0..k0+15
//                          lanes 16-31 row M=l-16: K k0+16..k0+31
// B fragment (32x16 f16): lanes 0-15 col N=l: K k0..k0+15
//                          lanes 16-31 col N=l-16: K k0+16..k0+31
// D (16x16 f32): vgpr j -> row j (lanes 0-15) / row j+8 (lanes 16-31), col lane%16

template <int K>
__global__ void gemm_wmma_k(const float* __restrict__ X, const float* __restrict__ W,
                            const float* __restrict__ b0, const float* __restrict__ b1,
                            float* __restrict__ Y, int N) {
  const int lane = threadIdx.x & 31;
  const int wv   = threadIdx.x >> 5;
  const int lr   = lane & 15;
  const int hi   = lane >> 4;
  const int row0 = blockIdx.y * 128 + wv * 16;
  const int col0 = blockIdx.x * 32;

  const float* arow  = X + (size_t)(row0 + lr) * K + hi * 16;
  const float* wrow0 = W + (size_t)(col0 + lr) * K + hi * 16;
  const float* wrow1 = wrow0 + (size_t)16 * K;

  __builtin_prefetch(arow, 0, 1);    // global_prefetch_b8
  __builtin_prefetch(wrow1, 0, 1);

  v8f acc0 = {}, acc1 = {};
#pragma unroll
  for (int k0 = 0; k0 < K; k0 += 32) {
    v16h a, p, q;
#pragma unroll
    for (int j = 0; j < 16; ++j) {
      a[j]     = (_Float16)arow[k0 + j];
    }
#pragma unroll
    for (int j = 0; j < 16; ++j) {
      p[j] = (_Float16)wrow0[k0 + j];
      q[j] = (_Float16)wrow1[k0 + j];
    }
    acc0 = __builtin_amdgcn_wmma_f32_16x16x32_f16(false, a, false, p,
                                                  (short)0, acc0, false, false);
    acc1 = __builtin_amdgcn_wmma_f32_16x16x32_f16(false, a, false, q,
                                                  (short)0, acc1, false, false);
  }

  float bias0 = 0.f, bias1 = 0.f;
  if (b0) { bias0 += b0[col0 + lr]; bias1 += b0[col0 + 16 + lr]; }
  if (b1) { bias0 += b1[col0 + lr]; bias1 += b1[col0 + 16 + lr]; }
  float* yp = Y + (size_t)(row0 + hi * 8) * N + col0 + lr;
#pragma unroll
  for (int j = 0; j < 8; ++j) {
    yp[(size_t)j * N]      = acc0[j] + bias0;
    yp[(size_t)j * N + 16] = acc1[j] + bias1;
  }
}

// ---------------- LSTM scan (one block per (batch); dir=0 fw / 1 bw) -------
// xg: [B,T,4H] precomputed x@Wih^T + bih + bhh ; Whh: [4H,H] ; out: [B,T,2H]
__global__ void lstm_scan_k(const float* __restrict__ xg, const float* __restrict__ Whh,
                            float* __restrict__ out, int H, int dir) {
  __shared__ float h[128], c[128], g[512];
  const int b = blockIdx.x;
  const int tid = threadIdx.x;          // 0 .. 4H-1
  if (tid < H) { h[tid] = 0.f; c[tid] = 0.f; }
  __syncthreads();
  const float* wr = Whh + (size_t)tid * H;
  for (int s = 0; s < T_; ++s) {
    const int t = dir ? (T_ - 1 - s) : s;
    const float* xr = xg + ((size_t)b * T_ + t) * (size_t)(4 * H);
    float acc = xr[tid];
    for (int j = 0; j < H; ++j) acc += wr[j] * h[j];
    g[tid] = acc;
    __syncthreads();
    if (tid < H) {
      float ig = g[tid], fg = g[H + tid], gg = g[2 * H + tid], og = g[3 * H + tid];
      float cn = sigm(fg) * c[tid] + sigm(ig) * tanhf(gg);
      float hn = sigm(og) * tanhf(cn);
      c[tid] = cn;
      h[tid] = hn;
      out[((size_t)b * T_ + t) * (size_t)(2 * H) + (size_t)dir * H + tid] = hn;
    }
    __syncthreads();
  }
}

// ---------------- EMA scan: grid=B, block=L --------------------------------
__global__ void ema_k(const float* __restrict__ x, const float* __restrict__ logit,
                      float* __restrict__ y) {
  const int b = blockIdx.x, f = threadIdx.x;
  const float a = sigm(logit[f]);
  size_t base = ((size_t)b * T_) * L_ + f;
  float yp = x[base];
  y[base] = yp;
  for (int t = 1; t < T_; ++t) {
    size_t off = base + (size_t)t * L_;
    yp = a * yp + (1.f - a) * x[off];
    y[off] = yp;
  }
}

// ---------------- misc elementwise -----------------------------------------
__global__ void hcat_k(const float* __restrict__ h2, const float* __restrict__ s,
                       const float* __restrict__ gate, float* __restrict__ out) {
  const long n = (long)M_ * 192;
  long i = (long)blockIdx.x * blockDim.x + threadIdx.x;
  if (i >= n) return;
  int c = (int)(i % 192);
  long m = i / 192;
  out[i] = (c < 128) ? h2[m * 128 + c] : sigm(gate[0]) * s[m * 64 + (c - 128)];
}

__global__ void sp_sqrt_k(const float* __restrict__ x, float* __restrict__ y,
                          long n, float eps) {
  long i = (long)blockIdx.x * blockDim.x + threadIdx.x;
  if (i < n) y[i] = sqrtf(softplusf(x[i]) + eps);
}

__global__ void ew_sub_k(const float* __restrict__ a, const float* __restrict__ b,
                         float* __restrict__ c, long n) {
  long i = (long)blockIdx.x * blockDim.x + threadIdx.x;
  if (i < n) c[i] = a[i] - b[i];
}

__global__ void first_diff_k(const float* __restrict__ x, float* __restrict__ y) {
  const long n = (long)M_ * L_;
  long i = (long)blockIdx.x * blockDim.x + threadIdx.x;
  if (i >= n) return;
  long t = (i / L_) % T_;
  y[i] = (t == 0) ? 0.f : x[i] - x[i - L_];
}

__global__ void gelu_k(float* __restrict__ x, long n) {
  long i = (long)blockIdx.x * blockDim.x + threadIdx.x;
  if (i < n) {
    float v = x[i];
    x[i] = 0.5f * v * (1.f + erff(v * 0.70710678118654752f));
  }
}

// ---------------- row-wise (block=64 per row) ------------------------------
__global__ void diff_normalize_k(const float* __restrict__ x, float* __restrict__ y) {
  __shared__ float red[64];
  const long m = blockIdx.x;
  const int f = threadIdx.x;
  const int t = (int)(m % T_);
  float v = (t > 0) ? (x[m * L_ + f] - x[(m - 1) * L_ + f]) : 0.f;
  float nrm = sqrtf(rowsum64(v * v, red));
  y[m * L_ + f] = v / fmaxf(nrm, 1e-12f);
}

__global__ void row_normalize_k(float* __restrict__ x) {
  __shared__ float red[64];
  const long m = blockIdx.x;
  const int f = threadIdx.x;
  float v = x[m * L_ + f];
  float nrm = sqrtf(rowsum64(v * v, red));
  x[m * L_ + f] = v / fmaxf(nrm, 1e-12f);
}

__global__ void merge_ln1_k(const float* __restrict__ oa, const float* __restrict__ ob,
                            const float* __restrict__ zp, const float* __restrict__ gmw,
                            const float* __restrict__ gmb, const float* __restrict__ gain,
                            const float* __restrict__ lng, const float* __restrict__ lnb,
                            float* __restrict__ out) {
  __shared__ float red[64];
  const long m = blockIdx.x;
  const int f = threadIdx.x;
  const long i = m * L_ + f;
  float a = oa[i], b = ob[i], z = zp[i];
  float wa = rowsum64(a * a, red) * (1.f / L_);
  float wb = rowsum64(b * b, red) * (1.f / L_);
  float gg = sigm(wa * gmw[0] + wb * gmw[1] + gmb[0]);
  float ao = gain[0] * (gg * a + (1.f - gg) * b);
  float ra = sqrtf(rowsum64(ao * ao, red) * (1.f / L_) + 1e-8f);
  float rz = sqrtf(rowsum64(z * z, red) * (1.f / L_) + 1e-8f);
  float scale = ra / (rz + 1e-8f);
  float yv = z * scale + ao;
  float mean = rowsum64(yv, red) * (1.f / L_);
  float d = yv - mean;
  float var = rowsum64(d * d, red) * (1.f / L_);
  out[i] = d * rsqrtf(var + 1e-5f) * lng[f] + lnb[f];
}

__global__ void add_scale_ln_k(const float* __restrict__ a, const float* __restrict__ fb,
                               const float* __restrict__ sc, const float* __restrict__ lng,
                               const float* __restrict__ lnb, float* __restrict__ out) {
  __shared__ float red[64];
  const long m = blockIdx.x;
  const int f = threadIdx.x;
  const long i = m * L_ + f;
  float yv = a[i] + sc[0] * fb[i];
  float mean = rowsum64(yv, red) * (1.f / L_);
  float d = yv - mean;
  float var = rowsum64(d * d, red) * (1.f / L_);
  out[i] = d * rsqrtf(var + 1e-5f) * lng[f] + lnb[f];
}

// ---------------- attention core (per b,h; online softmax) -----------------
// q,k,v,o: [B,T,64]; head h uses feature slice [h*8, h*8+8)
__global__ void attn_core_k(const float* __restrict__ q, const float* __restrict__ k,
                            const float* __restrict__ v, float* __restrict__ o) {
  const int HD = 8, NH = 8, D = 64;
  const int b = blockIdx.x / NH, h = blockIdx.x % NH;
  const int qi = blockIdx.y * blockDim.x + threadIdx.x;
  const size_t base = ((size_t)b * T_) * D + h * HD;
  float qv[HD];
#pragma unroll
  for (int d = 0; d < HD; ++d) qv[d] = q[base + (size_t)qi * D + d];
  const float scl = 0.3535533905932738f;  // 1/sqrt(8)
  float mx = -1e30f, l = 0.f, acc[HD] = {0.f, 0.f, 0.f, 0.f, 0.f, 0.f, 0.f, 0.f};
  __shared__ float ks[64][8];
  __shared__ float vs[64][8];
  for (int kt = 0; kt < T_; kt += 64) {
    __syncthreads();
    for (int i = threadIdx.x; i < 64 * HD; i += blockDim.x) {
      int r = i / HD, d = i % HD;
      ks[r][d] = k[base + (size_t)(kt + r) * D + d];
      vs[r][d] = v[base + (size_t)(kt + r) * D + d];
    }
    __syncthreads();
    for (int j = 0; j < 64; ++j) {
      float s = 0.f;
#pragma unroll
      for (int d = 0; d < HD; ++d) s += qv[d] * ks[j][d];
      s *= scl;
      if (s > mx) {
        float corr = __expf(mx - s);
        l *= corr;
#pragma unroll
        for (int d = 0; d < HD; ++d) acc[d] *= corr;
        mx = s;
      }
      float p = __expf(s - mx);
      l += p;
#pragma unroll
      for (int d = 0; d < HD; ++d) acc[d] += p * vs[j][d];
    }
  }
  float inv = 1.f / l;
#pragma unroll
  for (int d = 0; d < HD; ++d) o[base + (size_t)qi * D + d] = acc[d] * inv;
}

// ---------------- mixture gate + mu_base -----------------------------------
__global__ void gate_mu_k(const float* __restrict__ gl, const float* __restrict__ mk,
                          float* __restrict__ probs, float* __restrict__ mub) {
  const long n = (long)M_ * F_;
  long i = (long)blockIdx.x * blockDim.x + threadIdx.x;
  if (i >= n) return;
  const float* g = gl + i * 4;
  const float* mp = mk + i * 4;
  float g0 = g[0], g1 = g[1], g2 = g[2], g3 = g[3];
  float gm = 0.25f * (g0 + g1 + g2 + g3);
  float d0 = g0 - gm, d1 = g1 - gm, d2 = g2 - gm, d3 = g3 - gm;
  float gs = sqrtf((d0 * d0 + d1 * d1 + d2 * d2 + d3 * d3) * (1.f / 3.f));
  float inv = 1.f / (gs + 1e-5f);
  float z0 = d0 * inv, z1 = d1 * inv, z2 = d2 * inv, z3 = d3 * inv;
  float mx = fmaxf(fmaxf(z0, z1), fmaxf(z2, z3));
  float e0 = __expf(z0 - mx), e1 = __expf(z1 - mx), e2 = __expf(z2 - mx), e3 = __expf(z3 - mx);
  float r = 1.f / (e0 + e1 + e2 + e3);
  float p0 = e0 * r, p1 = e1 * r, p2 = e2 * r, p3 = e3 * r;
  probs[i * 4 + 0] = p0; probs[i * 4 + 1] = p1;
  probs[i * 4 + 2] = p2; probs[i * 4 + 3] = p3;
  mub[i] = mp[0] * p0 + mp[1] * p1 + mp[2] * p2 + mp[3] * p3;
}

// ---------------- event head + rec_mu --------------------------------------
__global__ void event_k(const float* __restrict__ evr, const float* __restrict__ mub,
                        const float* __restrict__ taulog, const float* __restrict__ egate,
                        const float* __restrict__ gamma, float* __restrict__ evres,
                        float* __restrict__ recmu) {
  __shared__ float red[64];
  const long m = blockIdx.x;
  const int f = threadIdx.x;
  const long i = m * L_ + f;
  float e = evr[i], mb = mub[i];
  float r1 = sqrtf(rowsum64(e * e, red) * (1.f / L_) + 1e-8f);
  float r2 = sqrtf(rowsum64(mb * mb, red) * (1.f / L_) + 1e-8f);
  float en = e / (r1 + 1e-8f);
  float tau = softplusf(taulog[f]);
  float mag = fabsf(en) - tau;
  float sh = (mag > 0.f) ? copysignf(mag, en) : 0.f;
  float res = sigm(egate[0]) * (sh * r2 * __expf(gamma[f]));
  evres[i] = res;
  recmu[i] = mb + res;
}

__global__ void rec_std_k(const float* __restrict__ lv, float* __restrict__ y) {
  int f = threadIdx.x;
  y[f] = sqrtf(softplusf(lv[f]) + 1e-4f);
}

// ---------------- input index map (setup_inputs flatten order) -------------
enum {
  IN_X = 0,
  E1F_WIH, E1F_WHH, E1F_BIH, E1F_BHH, E1B_WIH, E1B_WHH, E1B_BIH, E1B_BHH,   // 1..8
  E2F_WIH, E2F_WHH, E2F_BIH, E2F_BHH, E2B_WIH, E2B_WHH, E2B_BIH, E2B_BHH,   // 9..16
  D1F_WIH, D1F_WHH, D1F_BIH, D1F_BHH, D1B_WIH, D1B_WHH, D1B_BIH, D1B_BHH,   // 17..24
  D2F_WIH, D2F_WHH, D2F_BIH, D2F_BHH, D2B_WIH, D2B_WHH, D2B_BIH, D2B_BHH,   // 25..32
  INJ_W, INJ_GATE, TOMEAN_W, TOMEAN_B, TOLOGVAR_W, TOLOGVAR_B,              // 33..38
  PROJZ_W, PROJZ_B, PROJENC_W, PROJENC_B, EMA_Z, EMA_ENC,                   // 39..44
  AA_WQ, AA_WK, AA_WV, AA_WO, AA_BO,                                        // 45..49
  AB_WQ, AB_WK, AB_WV, AB_WO, AB_BO,                                        // 50..54
  ATTN_GAIN, GM_W, GM_B, LN1_G, LN1_B,                                      // 55..59
  FFN_W1, FFN_B1, FFN_W2, FFN_B2, LN2_G, LN2_B, FFN_SCALE,                  // 60..66
  DMK_W, DMK_B, DEC_LOGVAR, DG_W, DG_B, EH_W, SHRINK_LT, EVENT_GATE, EV_GAMMA // 67..75
};

// ---------------- host orchestration ---------------------------------------
static inline void gemm(hipStream_t st, const float* X, const float* W,
                        const float* b0, const float* b1, float* Y, int N, int K) {
  dim3 g(N / 32, M_ / 128);
  switch (K) {
    case 64:  gemm_wmma_k<64><<<g, 256, 0, st>>>(X, W, b0, b1, Y, N);  break;
    case 128: gemm_wmma_k<128><<<g, 256, 0, st>>>(X, W, b0, b1, Y, N); break;
    case 192: gemm_wmma_k<192><<<g, 256, 0, st>>>(X, W, b0, b1, Y, N); break;
    case 256: gemm_wmma_k<256><<<g, 256, 0, st>>>(X, W, b0, b1, Y, N); break;
    default:  break;  // all GEMMs in this model use K in {64,128,192,256}
  }
}

extern "C" void kernel_launch(void* const* d_in, const int* in_sizes, int n_in,
                              void* d_out, int out_size, void* d_ws, size_t ws_size,
                              hipStream_t stream) {
  (void)in_sizes; (void)n_in; (void)out_size; (void)ws_size;
  auto P = [&](int i) { return (const float*)d_in[i]; };
  float* out = (float*)d_out;
  float* ws  = (float*)d_ws;

  // workspace arena (floats)
  float* XG0 = ws;                         // M*512
  float* XG1 = XG0 + (size_t)M_ * 512;     // M*512
  float* H1  = XG1 + (size_t)M_ * 512;     // M*256 (enc_l1 out; later hd2)
  float* H2  = H1  + (size_t)M_ * 256;     // M*128 (enc_l2 out; later hd1)
  float* FB  = H2  + (size_t)M_ * 128;     // M*192 (h_cat, later ffn hidden)
  float* A0  = FB  + (size_t)M_ * 192;     // 21 x [M,64] buffers
  const size_t V64 = (size_t)M_ * 64;
  float *bS   = A0 + 0 * V64, *bLV  = A0 + 1 * V64, *bZP  = A0 + 2 * V64;
  float *bEP  = A0 + 3 * V64, *bESM = A0 + 4 * V64, *bZSM = A0 + 5 * V64;
  float *bQA  = A0 + 6 * V64, *bQ   = A0 + 7 * V64, *bK   = A0 + 8 * V64;
  float *bV   = A0 + 9 * V64, *bO   = A0 + 10 * V64, *bOA = A0 + 11 * V64;
  float *bOB  = A0 + 12 * V64, *bDE = A0 + 13 * V64, *bDZ = A0 + 14 * V64;
  float *bO1  = A0 + 15 * V64, *bFF = A0 + 16 * V64, *bO2 = A0 + 17 * V64;
  float *bMB  = A0 + 18 * V64, *bZS = A0 + 19 * V64, *bEV = A0 + 20 * V64;

  // output layout: rec_mu | rec_std(64) | mu | std | gate_probs | ev_res
  float* o_rec_mu  = out;
  float* o_rec_std = o_rec_mu + V64;
  float* o_mu      = o_rec_std + 64;
  float* o_std     = o_mu + V64;
  float* o_gp      = o_std + V64;
  float* o_ev      = o_gp + (size_t)M_ * 256;

  const long nML  = (long)M_ * 64;
  const long n192 = (long)M_ * 192;
  dim3 ewML((unsigned)((nML + 255) / 256)), ew192((unsigned)((n192 + 255) / 256));

  // ---- encoder BiLSTM 1 (F=64 -> H=128) ----
  gemm(stream, P(IN_X), P(E1F_WIH), P(E1F_BIH), P(E1F_BHH), XG0, 512, 64);
  gemm(stream, P(IN_X), P(E1B_WIH), P(E1B_BIH), P(E1B_BHH), XG1, 512, 64);
  lstm_scan_k<<<B_, 512, 0, stream>>>(XG0, P(E1F_WHH), H1, 128, 0);
  lstm_scan_k<<<B_, 512, 0, stream>>>(XG1, P(E1B_WHH), H1, 128, 1);

  // ---- encoder BiLSTM 2 (256 -> H=64) ----
  gemm(stream, H1, P(E2F_WIH), P(E2F_BIH), P(E2F_BHH), XG0, 256, 256);
  gemm(stream, H1, P(E2B_WIH), P(E2B_BIH), P(E2B_BHH), XG1, 256, 256);
  lstm_scan_k<<<B_, 256, 0, stream>>>(XG0, P(E2F_WHH), H2, 64, 0);
  lstm_scan_k<<<B_, 256, 0, stream>>>(XG1, P(E2B_WHH), H2, 64, 1);

  // ---- latent heads ----
  gemm(stream, P(IN_X), P(INJ_W), nullptr, nullptr, bS, 64, 64);
  hcat_k<<<ew192, 256, 0, stream>>>(H2, bS, P(INJ_GATE), FB);
  gemm(stream, FB, P(TOMEAN_W), P(TOMEAN_B), nullptr, o_mu, 64, 192);
  gemm(stream, FB, P(TOLOGVAR_W), P(TOLOGVAR_B), nullptr, bLV, 64, 192);
  sp_sqrt_k<<<ewML, 256, 0, stream>>>(bLV, o_std, nML, 1e-8f);
  gemm(stream, o_mu, P(PROJZ_W), P(PROJZ_B), nullptr, bZP, 64, 64);
  gemm(stream, H2, P(PROJENC_W), P(PROJENC_B), nullptr, bEP, 64, 128);

  // ---- EMA smoothing ----
  ema_k<<<B_, 64, 0, stream>>>(bEP, P(EMA_ENC), bESM);
  ema_k<<<B_, 64, 0, stream>>>(bZP, P(EMA_Z), bZSM);

  // ---- attention A (grad path) ----
  diff_normalize_k<<<M_, 64, 0, stream>>>(bESM, bQA);
  gemm(stream, bQA, P(AA_WQ), nullptr, nullptr, bQ, 64, 64);
  gemm(stream, bQA, P(AA_WK), nullptr, nullptr, bK, 64, 64);
  gemm(stream, bZSM, P(AA_WV), nullptr, nullptr, bV, 64, 64);
  attn_core_k<<<dim3(B_ * 8, T_ / 128), 128, 0, stream>>>(bQ, bK, bV, bO);
  gemm(stream, bO, P(AA_WO), P(AA_BO), nullptr, bOA, 64, 64);

  // ---- attention B (deviation path) ----
  ew_sub_k<<<ewML, 256, 0, stream>>>(bEP, bESM, bDE, nML);
  ew_sub_k<<<ewML, 256, 0, stream>>>(bZP, bZSM, bDZ, nML);
  row_normalize_k<<<M_, 64, 0, stream>>>(bDE);
  gemm(stream, bDE, P(AB_WQ), nullptr, nullptr, bQ, 64, 64);
  gemm(stream, bDE, P(AB_WK), nullptr, nullptr, bK, 64, 64);
  gemm(stream, bDZ, P(AB_WV), nullptr, nullptr, bV, 64, 64);
  attn_core_k<<<dim3(B_ * 8, T_ / 128), 128, 0, stream>>>(bQ, bK, bV, bO);
  gemm(stream, bO, P(AB_WO), P(AB_BO), nullptr, bOB, 64, 64);

  // ---- merge + LN1 + FFN + LN2 ----
  merge_ln1_k<<<M_, 64, 0, stream>>>(bOA, bOB, bZP, P(GM_W), P(GM_B), P(ATTN_GAIN),
                                     P(LN1_G), P(LN1_B), bO1);
  gemm(stream, bO1, P(FFN_W1), P(FFN_B1), nullptr, FB, 192, 64);
  gelu_k<<<ew192, 256, 0, stream>>>(FB, n192);
  gemm(stream, FB, P(FFN_W2), P(FFN_B2), nullptr, bFF, 64, 192);
  add_scale_ln_k<<<M_, 64, 0, stream>>>(bO1, bFF, P(FFN_SCALE), P(LN2_G), P(LN2_B), bO2);

  // ---- decoder BiLSTM 1 (64 -> H=64) ----
  gemm(stream, bO2, P(D1F_WIH), P(D1F_BIH), P(D1F_BHH), XG0, 256, 64);
  gemm(stream, bO2, P(D1B_WIH), P(D1B_BIH), P(D1B_BHH), XG1, 256, 64);
  lstm_scan_k<<<B_, 256, 0, stream>>>(XG0, P(D1F_WHH), H2, 64, 0);   // hd1 -> H2
  lstm_scan_k<<<B_, 256, 0, stream>>>(XG1, P(D1B_WHH), H2, 64, 1);

  // ---- decoder BiLSTM 2 (128 -> H=128) ----
  gemm(stream, H2, P(D2F_WIH), P(D2F_BIH), P(D2F_BHH), XG0, 512, 128);
  gemm(stream, H2, P(D2B_WIH), P(D2B_BIH), P(D2B_BHH), XG1, 512, 128);
  lstm_scan_k<<<B_, 512, 0, stream>>>(XG0, P(D2F_WHH), H1, 128, 0);  // hd2 -> H1
  lstm_scan_k<<<B_, 512, 0, stream>>>(XG1, P(D2B_WHH), H1, 128, 1);

  // ---- mixture head ----
  gemm(stream, H1, P(DMK_W), P(DMK_B), nullptr, XG0, 256, 256);      // mu_k
  gemm(stream, bZP, P(DG_W), P(DG_B), nullptr, XG1, 256, 64);        // gate logits
  gate_mu_k<<<ewML, 256, 0, stream>>>(XG1, XG0, o_gp, bMB);

  // ---- event head + outputs ----
  first_diff_k<<<ewML, 256, 0, stream>>>(bZP, bZS);
  gemm(stream, bZS, P(EH_W), nullptr, nullptr, bEV, 64, 64);
  event_k<<<M_, 64, 0, stream>>>(bEV, bMB, P(SHRINK_LT), P(EVENT_GATE), P(EV_GAMMA),
                                 o_ev, o_rec_mu);
  rec_std_k<<<1, 64, 0, stream>>>(P(DEC_LOGVAR), o_rec_std);
}
